// RQuadratic_8985071583510
// MI455X (gfx1250) — matrix operand order repaired and checked by reading, not compiled, not run on hardware
//
#include <hip/hip_runtime.h>

typedef __attribute__((ext_vector_type(2)))  float  v2f;
typedef __attribute__((ext_vector_type(4)))  float  v4f;
typedef __attribute__((ext_vector_type(8)))  float  v8f;
typedef __attribute__((ext_vector_type(16))) __bf16 v16bf;

#define BATCH   32768
#define D       64
#define D2      4096              // D*D
#define TB      16                // batch tile per workgroup (WMMA M dim)
#define THREADS 512               // 16 waves of 32
#define LDS_FLOATS (TB*D2 + TB*D + TB*D)   // M tile + g tile + v tile

// ---------------------------------------------------------------------------
// Prep: split+transpose weight rows 0..63 into bf16 hi/lo, layout [c][k]
// whi/wlo: [4096][64] bf16 (512 KB each) in d_ws.
// ---------------------------------------------------------------------------
__global__ __launch_bounds__(256)
void wsplit_kernel(const float* __restrict__ w,
                   __bf16* __restrict__ whi, __bf16* __restrict__ wlo)
{
    const int t = blockIdx.x * 256 + threadIdx.x;   // 0 .. 64*4096-1
    const int c = t & (D2 - 1);                      // coalesced read dim
    const int k = t >> 12;                           // 0..63
    const float f = w[(size_t)k * D2 + c];
    const __bf16 h = (__bf16)f;                      // RNE
    const __bf16 l = (__bf16)(f - (float)h);         // residual, RNE
    whi[(size_t)c * D + k] = h;
    wlo[(size_t)c * D + k] = l;
}

// ---------------------------------------------------------------------------
// Main: out_b = (1/8) * M_b * (M_b^T g_b),  M_b = [x_b,1] @ W  (64x64)
// Big GEMM done as bf16x3 split WMMA (fp32-accurate), M tile lives in LDS.
// ---------------------------------------------------------------------------
__global__ __launch_bounds__(THREADS)
void rq_kernel(const float* __restrict__ x,
               const float* __restrict__ gH,
               const float* __restrict__ w,
               const __bf16* __restrict__ whi,
               const __bf16* __restrict__ wlo,
               float* __restrict__ out)
{
    extern __shared__ float lds[];
    float* Mlds = lds;                 // [TB][D2]  256 KB
    float* glds = lds + TB * D2;       // [TB][D]
    float* vlds = glds + TB * D;       // [TB][D]

    const int tid  = threadIdx.x;
    const int lane = tid & 31;
    const int wave = tid >> 5;
    const int half = lane >> 4;        // 0 or 1
    const int l16  = lane & 15;
    const int b0   = blockIdx.x * TB;

    // ---- stage grad_H tile into LDS (1024 floats, 2 per thread) ----
    glds[tid]       = gH[b0 * D + tid];
    glds[tid + 512] = gH[b0 * D + tid + 512];

    // ---- build bf16 hi/lo A fragments for both K-chunks (once per WG) ----
    // A 16x32 fragment, contiguous k order (same as B): lane(h,l16) element e of v16bf holds
    //   K = kc*32 + 16h + e,  M = l16
    v16bf ahi[2], alo[2];
    const float* xrow = x + (size_t)(b0 + l16) * D;
    #pragma unroll
    for (int kc = 0; kc < 2; ++kc) {
        float buf[16];
        *(v4f*)(buf + 0)  = *(const v4f*)(xrow + kc * 32 + 16 * half + 0);
        *(v4f*)(buf + 4)  = *(const v4f*)(xrow + kc * 32 + 16 * half + 4);
        *(v4f*)(buf + 8)  = *(const v4f*)(xrow + kc * 32 + 16 * half + 8);
        *(v4f*)(buf + 12) = *(const v4f*)(xrow + kc * 32 + 16 * half + 12);
        #pragma unroll
        for (int e = 0; e < 16; ++e) {
            const float f = buf[e];
            const __bf16 h = (__bf16)f;
            ahi[kc][e] = h;
            alo[kc][e] = (__bf16)(f - (float)h);
        }
    }

    // ---- big GEMM: M[16 batches][4096 cols], K=64, bias folded into C ----
    const float* wbias = w + (size_t)D * D2;   // weight row 64 = bias term
    #pragma unroll 1
    for (int ct = 0; ct < 16; ++ct) {
        const int c0 = wave * 256 + ct * 16;   // this wave's 16-column tile
        const float bv = wbias[c0 + l16];
        v8f acc;
        #pragma unroll
        for (int r = 0; r < 8; ++r) acc[r] = bv;   // C[m,n] = bias[n]

        // B 32x16 bf16 fragment, same k order as A: lane(h,l16) holds K = kc*32 + 16h .. +15
        // consecutively -> two 32B contiguous loads from transposed split W.
        const __bf16* bh_row = whi + (size_t)(c0 + l16) * D + 16 * half;
        const __bf16* bl_row = wlo + (size_t)(c0 + l16) * D + 16 * half;
        #pragma unroll
        for (int kc = 0; kc < 2; ++kc) {
            const v16bf bh = *(const v16bf*)(bh_row + kc * 32);
            const v16bf bl = *(const v16bf*)(bl_row + kc * 32);
            // x*w ~= xh*wh + xh*wl + xl*wh   (xl*wl ~ 2^-17, dropped)
            acc = __builtin_amdgcn_wmma_f32_16x16x32_bf16(
                      false, ahi[kc], false, bh, (short)0, acc, false, false);
            acc = __builtin_amdgcn_wmma_f32_16x16x32_bf16(
                      false, ahi[kc], false, bl, (short)0, acc, false, false);
            acc = __builtin_amdgcn_wmma_f32_16x16x32_bf16(
                      false, alo[kc], false, bh, (short)0, acc, false, false);
        }
        // C layout: reg r holds row m = r + 8*half, col n = l16
        #pragma unroll
        for (int r = 0; r < 8; ++r)
            Mlds[(r + 8 * half) * D2 + c0 + l16] = acc[r];
    }
    __syncthreads();

    // ---- stage 2a: v[b][k] = sum_i g[b][i] * M[b][i*64 + k] ----
    for (int t = tid; t < TB * D; t += THREADS) {
        const int b = t >> 6, k = t & 63;
        const float* Mb = Mlds + b * D2;
        const float* gb = glds + b * D;
        float s = 0.f;
        #pragma unroll 8
        for (int i = 0; i < D; ++i)
            s += gb[i] * Mb[i * D + k];
        vlds[t] = s;
    }
    __syncthreads();

    // ---- stage 2b: out[b][e] = (1/8) sum_k M[b][e*64+k] * v[b][k] ----
    for (int t = tid; t < TB * D; t += THREADS) {
        const int b = t >> 6, e = t & 63;
        const float* Mb = Mlds + b * D2 + e * D;
        const float* vb = vlds + b * D;
        float s = 0.f;
        #pragma unroll 8
        for (int kk = 0; kk < D; ++kk) {
            const int k = (kk + lane) & 63;
            s += Mb[k] * vb[k];
        }
        out[(size_t)(b0 + b) * D + e] = s * 0.125f;   // 1/sqrt(64)
    }
}

extern "C" void kernel_launch(void* const* d_in, const int* in_sizes, int n_in,
                              void* d_out, int out_size, void* d_ws, size_t ws_size,
                              hipStream_t stream) {
    (void)in_sizes; (void)n_in; (void)out_size; (void)ws_size;
    const float* x  = (const float*)d_in[0];
    const float* gH = (const float*)d_in[1];
    const float* w  = (const float*)d_in[2];
    float* out = (float*)d_out;

    // workspace: Whi_t [4096][64] bf16 (512KB) + Wlo_t (512KB) = 1 MB
    __bf16* whi = (__bf16*)d_ws;
    __bf16* wlo = whi + (size_t)D2 * D;

    wsplit_kernel<<<(D * D2) / 256, 256, 0, stream>>>(w, whi, wlo);

    const int lds_bytes = LDS_FLOATS * (int)sizeof(float);   // 270336 B
    hipFuncSetAttribute(reinterpret_cast<const void*>(rq_kernel),
                        hipFuncAttributeMaxDynamicSharedMemorySize, lds_bytes);
    rq_kernel<<<BATCH / TB, THREADS, lds_bytes, stream>>>(x, gH, w, whi, wlo, out);
}
